// GATPortfolioNet_65352222376031
// MI455X (gfx1250) — compile-verified, matrix operand order checked by reading, not run on hardware
//
#include <hip/hip_runtime.h>

// ---------------------------------------------------------------------------
// GATPortfolioNet on MI455X (gfx1250, wave32, WMMA).
// GEMMs use v_wmma_f32_16x16x32_f16 (f16 in, f32 accum). GRU recurrence is
// done fully in-wave (16 stocks per wave, h kept in LDS, whh staged in LDS).
// ---------------------------------------------------------------------------

#define N_STOCKS 2000
#define SEQ_LEN  256
#define N_FEAT   20
#define D_MODEL  128
#define MEM_LEN  12
#define GRU_H    64

typedef __attribute__((ext_vector_type(16))) _Float16 v16h;
typedef __attribute__((ext_vector_type(8)))  _Float16 v8h;
typedef __attribute__((ext_vector_type(8)))  float    v8f;
typedef _Float16 h16;

__device__ __forceinline__ float sigmoidf_(float x){ return 1.f/(1.f+__expf(-x)); }
__device__ __forceinline__ float geluf_(float x){ return 0.5f*x*(1.f+erff(x*0.70710678118654752f)); }
__device__ __forceinline__ unsigned f2ord(float f){ unsigned u=__float_as_uint(f); return (u&0x80000000u)? ~u : (u|0x80000000u); }
__device__ __forceinline__ float ord2f(unsigned u){ return __uint_as_float((u&0x80000000u)? (u^0x80000000u) : ~u); }

// Fragment loader for A (rows=M,K contiguous) and B (rows=Ncol,K contiguous).
// ISA 16-bit 16x32 layout: lanes 0-15 -> rows 0-15 with K {0..7,16..23};
// lanes 16-31 -> same rows with K {8..15,24..31}.
template<typename PT>
__device__ __forceinline__ v16h load_frag16(PT base, size_t ld){
  const int l = threadIdx.x & 31;
  PT p = base + (size_t)(l & 15)*ld + ((l >> 4) << 3);
  v8h lo = *(const v8h*)p;
  v8h hi = *(const v8h*)(p + 16);
  return __builtin_shufflevector(lo, hi, 0,1,2,3,4,5,6,7,8,9,10,11,12,13,14,15);
}

__device__ __forceinline__ v8f wmma_f16(v16h a, v16h b, v8f c){
  return __builtin_amdgcn_wmma_f32_16x16x32_f16(false, a, false, b, (short)0, c, false, false);
}

// ------------------------------- utility kernels ---------------------------

__global__ void f32_to_f16_kernel(const float* __restrict__ in, h16* __restrict__ out, size_t n){
  size_t i = blockIdx.x*(size_t)blockDim.x + threadIdx.x;
  if (i < n) out[i] = (h16)in[i];
}

__global__ void fill_u32_kernel(unsigned* __restrict__ p, unsigned v, size_t n){
  size_t i = blockIdx.x*(size_t)blockDim.x + threadIdx.x;
  if (i < n) p[i] = v;
}

// ------------------------- stage 1: input proj + LN + GELU -----------------

__global__ __launch_bounds__(128)
void in_proj_kernel(const float* __restrict__ x, const float* __restrict__ w,
                    const float* __restrict__ b, const float* __restrict__ lg,
                    const float* __restrict__ lb, h16* __restrict__ out)
{
  __shared__ float sx[N_FEAT];
  __shared__ float red[128];
  const size_t row = blockIdx.x;
  const int c = threadIdx.x;
  if (c < N_FEAT) sx[c] = x[row*N_FEAT + c];
  __syncthreads();
  float acc = b[c];
  #pragma unroll
  for (int k=0;k<N_FEAT;k++) acc += w[c*N_FEAT+k]*sx[k];
  red[c]=acc; __syncthreads();
  for (int s=64;s;s>>=1){ if (c<s) red[c]+=red[c+s]; __syncthreads(); }
  float mean = red[0]*(1.f/128.f); __syncthreads();
  float d = acc-mean; red[c]=d*d; __syncthreads();
  for (int s=64;s;s>>=1){ if (c<s) red[c]+=red[c+s]; __syncthreads(); }
  float var = red[0]*(1.f/128.f);
  float y = d*rsqrtf(var+1e-5f)*lg[c]+lb[c];
  out[row*D_MODEL + c] = (h16)geluf_(y);
}

// ------------------------- stage 2: in-wave GRU scan -----------------------
// Each wave owns 16 stocks. Per step: 48 WMMA for x@wih^T (B frags from
// global), 24 WMMA for h@whh^T (B frags from LDS). h tile lives in LDS f16.

__global__ __launch_bounds__(128)
void gru_scan_kernel(const h16* __restrict__ xin, const h16* __restrict__ wih,
                     const h16* __restrict__ whh, const float* __restrict__ bih,
                     const float* __restrict__ bhh, h16* __restrict__ yout,
                     int reverse, int colBase)
{
  __shared__ __attribute__((aligned(16))) h16 s_whh[192*GRU_H];
  __shared__ __attribute__((aligned(16))) h16 s_h[4][16*GRU_H];
  const int tid = threadIdx.x, wave = tid>>5, lane = tid&31;
  for (int i=tid; i<192*GRU_H; i+=128) s_whh[i] = whh[i];
  for (int i=lane; i<16*GRU_H; i+=32) s_h[wave][i] = (h16)0.f;

  const int nTiles = N_STOCKS/16;                 // 125
  const int tileIdx = blockIdx.x*4 + wave;
  const bool valid = tileIdx < nTiles;
  const int tile = valid ? tileIdx : nTiles-1;    // clamp so all waves hit barriers
  const size_t stock0 = (size_t)tile*16;
  const int nn = lane & 15;
  const int mhi = (lane >> 4) * 8;

  float brz[8], bin_[4], bhn[4];
  #pragma unroll
  for (int g=0; g<8; ++g) brz[g] = bih[g*16+nn] + bhh[g*16+nn];
  #pragma unroll
  for (int g=0; g<4; ++g){ bin_[g]=bih[128+g*16+nn]; bhn[g]=bhh[128+g*16+nn]; }
  h16* shh = s_h[wave];
  const v8f vzero = {};
  __syncthreads();

  for (int step=0; step<SEQ_LEN; ++step){
    __syncthreads();   // order h writes (prev step) vs frag reads (this step)
    const int t = reverse ? (SEQ_LEN-1-step) : step;
    v8f acc_rz[8], acc_in[4], acc_hn[4];
    #pragma unroll
    for (int i=0;i<8;i++) acc_rz[i]=vzero;
    #pragma unroll
    for (int i=0;i<4;i++){ acc_in[i]=vzero; acc_hn[i]=vzero; }

    const h16* xbase = xin + (stock0*SEQ_LEN + (size_t)t)*D_MODEL;
    #pragma unroll
    for (int ks=0; ks<4; ++ks){
      v16h a = load_frag16(xbase + ks*32, (size_t)SEQ_LEN*D_MODEL);
      #pragma unroll
      for (int nt_=0; nt_<12; ++nt_){
        v16h b = load_frag16(wih + (size_t)nt_*16*D_MODEL + ks*32, (size_t)D_MODEL);
        if (nt_ < 8) acc_rz[nt_]   = wmma_f16(a,b,acc_rz[nt_]);
        else         acc_in[nt_-8] = wmma_f16(a,b,acc_in[nt_-8]);
      }
    }
    #pragma unroll
    for (int ks=0; ks<2; ++ks){
      v16h a = load_frag16((const h16*)(shh + ks*32), (size_t)GRU_H);
      #pragma unroll
      for (int nt_=0; nt_<12; ++nt_){
        v16h b = load_frag16((const h16*)(s_whh + (size_t)nt_*16*GRU_H + ks*32), (size_t)GRU_H);
        if (nt_ < 8) acc_rz[nt_]   = wmma_f16(a,b,acc_rz[nt_]);
        else         acc_hn[nt_-8] = wmma_f16(a,b,acc_hn[nt_-8]);
      }
    }
    if (step+1 < SEQ_LEN){
      const int tn = reverse ? (SEQ_LEN-2-step) : step+1;
      __builtin_prefetch(xin + (stock0*SEQ_LEN + (size_t)tn)*D_MODEL, 0, 1);
    }

    // gates in C-fragment layout: vgpr r -> M = r + mhi, lane -> N = nn
    #pragma unroll
    for (int jt=0; jt<4; ++jt){
      #pragma unroll
      for (int r=0; r<8; ++r){
        float rg = sigmoidf_(acc_rz[jt][r]   + brz[jt]);
        float zg = sigmoidf_(acc_rz[4+jt][r] + brz[4+jt]);
        float ng = tanhf(acc_in[jt][r] + bin_[jt] + rg*(acc_hn[jt][r] + bhn[jt]));
        const int m = r + mhi, j = jt*16 + nn;
        float hp = (float)shh[m*GRU_H + j];
        float hnew = (1.f-zg)*ng + zg*hp;
        shh[m*GRU_H + j] = (h16)hnew;
        if (valid)
          yout[((stock0+(size_t)m)*SEQ_LEN + (size_t)t)*D_MODEL + colBase + j] = (h16)hnew;
      }
    }
  }
}

// ------------------------- generic WMMA linear: C = A @ W^T ----------------

__global__ __launch_bounds__(32)
void wmma_linear_kernel(const h16* __restrict__ A, const h16* __restrict__ W,
                        float* __restrict__ C, int Nc, int K)
{
  const size_t mt = blockIdx.x, nt = blockIdx.y;
  v8f acc = {};
  for (int k=0; k<K; k+=32){
    v16h a = load_frag16(A + mt*16*(size_t)K + k, (size_t)K);
    v16h b = load_frag16(W + nt*16*(size_t)K + k, (size_t)K);
    acc = wmma_f16(a,b,acc);
  }
  const int lane = threadIdx.x & 31;
  const int nn = lane & 15, mhi = (lane>>4)*8;
  #pragma unroll
  for (int r=0;r<8;r++){
    size_t row = mt*16 + (size_t)(r + mhi);
    C[row*(size_t)Nc + nt*16 + nn] = acc[r];
  }
}

// ------------------------- attention pooling over T ------------------------

__global__ __launch_bounds__(256)
void attnpool_kernel(const h16* __restrict__ inp, const float* __restrict__ aw,
                     const float* __restrict__ ab, h16* __restrict__ pooled)
{
  __shared__ float s_alpha[256];
  __shared__ float red[256];
  const int n = blockIdx.x, t = threadIdx.x;
  const h16* row = inp + ((size_t)n*SEQ_LEN + t)*D_MODEL;
  float acc = ab[0];
  #pragma unroll 4
  for (int c=0;c<D_MODEL;c++) acc += (float)row[c]*aw[c];
  red[t]=acc; __syncthreads();
  for (int s=128;s;s>>=1){ if (t<s) red[t]=fmaxf(red[t],red[t+s]); __syncthreads(); }
  float mx = red[0]; __syncthreads();
  float e = __expf(acc-mx);
  red[t]=e; __syncthreads();
  for (int s=128;s;s>>=1){ if (t<s) red[t]+=red[t+s]; __syncthreads(); }
  float inv = 1.f/red[0]; __syncthreads();
  s_alpha[t] = e*inv;
  __syncthreads();
  if (t < D_MODEL){
    float p = 0.f;
    for (int tt=0; tt<SEQ_LEN; ++tt)
      p += (float)inp[((size_t)n*SEQ_LEN+tt)*D_MODEL + t] * s_alpha[tt];
    pooled[(size_t)n*D_MODEL + t] = (h16)p;
  }
}

// ------------------------- bias+gelu+LN / add+LN ---------------------------

__global__ __launch_bounds__(128)
void gelu_ln_kernel(const float* __restrict__ in, const float* __restrict__ bias,
                    const float* __restrict__ g, const float* __restrict__ b,
                    float* __restrict__ outF, h16* __restrict__ outH)
{
  __shared__ float red[128];
  const int n=blockIdx.x, c=threadIdx.x;
  float v = geluf_(in[(size_t)n*D_MODEL+c] + bias[c]);
  red[c]=v; __syncthreads();
  for (int s=64;s;s>>=1){ if (c<s) red[c]+=red[c+s]; __syncthreads(); }
  float mean=red[0]*(1.f/128.f); __syncthreads();
  float d=v-mean; red[c]=d*d; __syncthreads();
  for (int s=64;s;s>>=1){ if (c<s) red[c]+=red[c+s]; __syncthreads(); }
  float var=red[0]*(1.f/128.f);
  float y=d*rsqrtf(var+1e-5f)*g[c]+b[c];
  outF[(size_t)n*D_MODEL+c]=y;
  outH[(size_t)n*D_MODEL+c]=(h16)y;
}

__global__ __launch_bounds__(128)
void add_ln_kernel(const float* __restrict__ a, const float* __restrict__ r,
                   const float* __restrict__ g, const float* __restrict__ b,
                   float* __restrict__ out)
{
  __shared__ float red[128];
  const int n=blockIdx.x, c=threadIdx.x;
  float v = a[(size_t)n*D_MODEL+c] + r[(size_t)n*D_MODEL+c];
  red[c]=v; __syncthreads();
  for (int s=64;s;s>>=1){ if (c<s) red[c]+=red[c+s]; __syncthreads(); }
  float mean=red[0]*(1.f/128.f); __syncthreads();
  float d=v-mean; red[c]=d*d; __syncthreads();
  for (int s=64;s;s>>=1){ if (c<s) red[c]+=red[c+s]; __syncthreads(); }
  float var=red[0]*(1.f/128.f);
  out[(size_t)n*D_MODEL+c]=d*rsqrtf(var+1e-5f)*g[c]+b[c];
}

// ------------------------------- GAT stages --------------------------------

__global__ __launch_bounds__(128)
void gat_attn_coef_kernel(const float* __restrict__ xl, const float* __restrict__ as_,
                          const float* __restrict__ ad_, float* __restrict__ asrc,
                          float* __restrict__ adst)
{
  const int n=blockIdx.x, h=threadIdx.x>>5, c=threadIdx.x&31;
  float v = xl[(size_t)n*D_MODEL + h*32 + c];
  float ps = v*as_[h*32+c], pd = v*ad_[h*32+c];
  #pragma unroll
  for (int o=16;o;o>>=1){ ps += __shfl_down(ps,o); pd += __shfl_down(pd,o); }
  if (c==0){ asrc[(size_t)n*4+h]=ps; adst[(size_t)n*4+h]=pd; }
}

__global__ void gat_edge_max_kernel(const int* __restrict__ ei, int E,
                                    const float* __restrict__ asrc, const float* __restrict__ adst,
                                    float* __restrict__ ebuf, unsigned* __restrict__ emaxu)
{
  const int idx = blockIdx.x*blockDim.x + threadIdx.x;
  const int tot = (E + N_STOCKS)*4;
  if (idx >= tot) return;
  const int e = idx >> 2, h = idx & 3;
  const int s = e < E ? ei[e]     : e - E;
  const int d = e < E ? ei[E + e] : e - E;
  float v = asrc[(size_t)s*4+h] + adst[(size_t)d*4+h];
  v = v > 0.f ? v : 0.2f*v;                 // leaky_relu 0.2
  ebuf[idx] = v;
  atomicMax(&emaxu[(size_t)d*4+h], f2ord(v));
}

__global__ void gat_edge_soft_kernel(const int* __restrict__ ei, int E,
                                     const float* __restrict__ ebuf, const unsigned* __restrict__ emaxu,
                                     float* __restrict__ eebuf, float* __restrict__ denom)
{
  const int idx = blockIdx.x*blockDim.x + threadIdx.x;
  const int tot = (E + N_STOCKS)*4;
  if (idx >= tot) return;
  const int e = idx >> 2, h = idx & 3;
  const int d = e < E ? ei[E + e] : e - E;
  float ee = __expf(ebuf[idx] - ord2f(emaxu[(size_t)d*4+h]));
  eebuf[idx] = ee;
  atomicAdd(&denom[(size_t)d*4+h], ee);
}

__global__ __launch_bounds__(128)
void gat_scatter_kernel(const int* __restrict__ ei, int E, const float* __restrict__ xl,
                        const float* __restrict__ eebuf, const float* __restrict__ denom,
                        float* __restrict__ agg)
{
  const int e = blockIdx.x, c = threadIdx.x, h = c >> 5;
  const int s = e < E ? ei[e]     : e - E;
  const int d = e < E ? ei[E + e] : e - E;
  const float alpha = eebuf[(size_t)e*4+h] / denom[(size_t)d*4+h];
  atomicAdd(&agg[(size_t)d*D_MODEL + c], xl[(size_t)s*D_MODEL + c]*alpha);
}

__global__ void gat_finish_kernel(const float* __restrict__ agg, const float* __restrict__ bias,
                                  float* __restrict__ hout, h16* __restrict__ houtH)
{
  const size_t idx = blockIdx.x*(size_t)blockDim.x + threadIdx.x;
  if (idx >= (size_t)N_STOCKS*D_MODEL) return;
  const int c = (int)(idx & (D_MODEL-1));
  float v = agg[idx] + bias[c];
  v = v > 0.f ? v : (__expf(v) - 1.f);      // elu
  hout[idx] = v;
  houtH[idx] = (h16)v;
}

// ---------------------------- portfolio memory -----------------------------

__global__ __launch_bounds__(128)
void mem_seq_kernel(const float* __restrict__ history, const float* __restrict__ w,
                    const float* __restrict__ b, float* __restrict__ seq)
{
  const int m = blockIdx.x, c = threadIdx.x;
  const float* hr = history + (size_t)m*2*N_STOCKS;
  const float* wr = w + (size_t)c*2*N_STOCKS;
  float acc = 0.f;
  for (int k=0; k<2*N_STOCKS; ++k) acc += hr[k]*wr[k];
  seq[(size_t)m*D_MODEL + c] = acc + b[c];
}

__global__ __launch_bounds__(128)
void mem_gru_kernel(const float* __restrict__ seq, const float* __restrict__ wih,
                    const float* __restrict__ whh, const float* __restrict__ bih,
                    const float* __restrict__ bhh, const float* __restrict__ g,
                    const float* __restrict__ b, float* __restrict__ regret)
{
  __shared__ float sh[128];
  __shared__ float red[128];
  const int j = threadIdx.x;
  sh[j] = 0.f; float h = 0.f;
  __syncthreads();
  for (int t=0; t<MEM_LEN; ++t){
    const float* xrow = seq + (size_t)t*D_MODEL;
    float ir=bih[j], iz=bih[128+j], in_=bih[256+j];
    float hr=bhh[j], hz=bhh[128+j], hn=bhh[256+j];
    for (int k=0;k<128;k++){
      float xv=xrow[k], hv=sh[k];
      ir  += wih[(size_t)j*128+k]*xv;
      iz  += wih[(size_t)(128+j)*128+k]*xv;
      in_ += wih[(size_t)(256+j)*128+k]*xv;
      hr  += whh[(size_t)j*128+k]*hv;
      hz  += whh[(size_t)(128+j)*128+k]*hv;
      hn  += whh[(size_t)(256+j)*128+k]*hv;
    }
    float r = sigmoidf_(ir+hr), z = sigmoidf_(iz+hz);
    float n_ = tanhf(in_ + r*hn);
    h = (1.f-z)*n_ + z*h;
    __syncthreads();
    sh[j] = h;
    __syncthreads();
  }
  red[j]=h; __syncthreads();
  for (int s=64;s;s>>=1){ if (j<s) red[j]+=red[j+s]; __syncthreads(); }
  float mean=red[0]*(1.f/128.f); __syncthreads();
  float d=h-mean; red[j]=d*d; __syncthreads();
  for (int s=64;s;s>>=1){ if (j<s) red[j]+=red[j+s]; __syncthreads(); }
  float var=red[0]*(1.f/128.f);
  regret[j] = d*rsqrtf(var+1e-5f)*g[j]+b[j];
}

// ------------------------------- head + softmax ----------------------------

__global__ void feat_kernel(const float* __restrict__ hf, const float* __restrict__ rg,
                            h16* __restrict__ feat)
{
  const size_t idx = blockIdx.x*(size_t)blockDim.x + threadIdx.x;
  if (idx >= (size_t)N_STOCKS*256) return;
  const int c = (int)(idx & 255);
  const size_t n = idx >> 8;
  float v = (c < 128) ? hf[n*D_MODEL + c] : rg[c-128];
  feat[idx] = (h16)v;
}

__global__ __launch_bounds__(128)
void head_kernel(const float* __restrict__ s1, const float* __restrict__ b1,
                 const float* __restrict__ w2, const float* __restrict__ b2,
                 float* __restrict__ scores)
{
  __shared__ float red[128];
  const int n=blockIdx.x, c=threadIdx.x;
  float v = geluf_(s1[(size_t)n*D_MODEL+c] + b1[c]) * w2[c];
  red[c]=v; __syncthreads();
  for (int s=64;s;s>>=1){ if (c<s) red[c]+=red[c+s]; __syncthreads(); }
  if (c==0) scores[n] = red[0] + b2[0];
}

__global__ __launch_bounds__(1024)
void softmax_project_kernel(const float* __restrict__ scores, const float* __restrict__ temp,
                            float* __restrict__ out)
{
  __shared__ float red[1024];
  const int t = threadIdx.x;
  const float ti = 1.f/fmaxf(temp[0], 0.1f);
  const bool va = t < N_STOCKS, vb = (t+1024) < N_STOCKS;
  float a = va ? scores[t]*ti       : -3.0e38f;
  float b = vb ? scores[t+1024]*ti  : -3.0e38f;
  red[t]=fmaxf(a,b); __syncthreads();
  for (int s=512;s;s>>=1){ if (t<s) red[t]=fmaxf(red[t],red[t+s]); __syncthreads(); }
  float mx=red[0]; __syncthreads();
  float ea = va ? __expf(a-mx) : 0.f;
  float eb = vb ? __expf(b-mx) : 0.f;
  red[t]=ea+eb; __syncthreads();
  for (int s=512;s;s>>=1){ if (t<s) red[t]+=red[t+s]; __syncthreads(); }
  float inv=1.f/red[0]; __syncthreads();
  float wa=ea*inv, wb=eb*inv;
  for (int it=0; it<10; ++it){
    wa=fminf(fmaxf(wa,0.02f),0.25f);
    wb=fminf(fmaxf(wb,0.02f),0.25f);
    red[t]=(va?wa:0.f)+(vb?wb:0.f); __syncthreads();
    for (int s=512;s;s>>=1){ if (t<s) red[t]+=red[t+s]; __syncthreads(); }
    float inv2=1.f/red[0]; __syncthreads();
    wa*=inv2; wb*=inv2;
  }
  if (va) out[t]=wa;
  if (vb) out[t+1024]=wb;
}

// ------------------------------- host driver -------------------------------

extern "C" void kernel_launch(void* const* d_in, const int* in_sizes, int n_in,
                              void* d_out, int out_size, void* d_ws, size_t ws_size,
                              hipStream_t stream)
{
  const float* x           = (const float*)d_in[0];
  const int*   edge_index  = (const int*)d_in[1];
  const float* history     = (const float*)d_in[2];
  const float* te_in_w     = (const float*)d_in[3];
  const float* te_in_b     = (const float*)d_in[4];
  const float* te_in_ln_g  = (const float*)d_in[5];
  const float* te_in_ln_b  = (const float*)d_in[6];
  const float* te_gru_wih  = (const float*)d_in[7];
  const float* te_gru_whh  = (const float*)d_in[8];
  const float* te_gru_bih  = (const float*)d_in[9];
  const float* te_gru_bhh  = (const float*)d_in[10];
  const float* te_attn_w   = (const float*)d_in[11];
  const float* te_attn_b   = (const float*)d_in[12];
  const float* te_out_w    = (const float*)d_in[13];
  const float* te_out_b    = (const float*)d_in[14];
  const float* te_ln_g     = (const float*)d_in[15];
  const float* te_ln_b     = (const float*)d_in[16];
  const float* gat_w       = (const float*)d_in[17];
  const float* gat_att_src = (const float*)d_in[18];
  const float* gat_att_dst = (const float*)d_in[19];
  const float* gat_bias    = (const float*)d_in[20];
  const float* gat_ln_g    = (const float*)d_in[21];
  const float* gat_ln_b    = (const float*)d_in[22];
  const float* mem_in_w    = (const float*)d_in[23];
  const float* mem_in_b    = (const float*)d_in[24];
  const float* mem_gru_wih = (const float*)d_in[25];
  const float* mem_gru_whh = (const float*)d_in[26];
  const float* mem_gru_bih = (const float*)d_in[27];
  const float* mem_gru_bhh = (const float*)d_in[28];
  const float* mem_ln_g    = (const float*)d_in[29];
  const float* mem_ln_b    = (const float*)d_in[30];
  const float* head_w1     = (const float*)d_in[31];
  const float* head_b1     = (const float*)d_in[32];
  const float* head_w2     = (const float*)d_in[33];
  const float* head_b2     = (const float*)d_in[34];
  const float* temperature = (const float*)d_in[35];
  (void)n_in; (void)out_size; (void)ws_size;

  const int E  = in_sizes[1] / 2;
  const int E2 = E + N_STOCKS;
  const size_t NT = (size_t)N_STOCKS * SEQ_LEN;

  char* ws = (char*)d_ws; size_t off = 0;
  auto alloc = [&](size_t bytes)->char* {
    char* p = ws + off; off = (off + bytes + 255) & ~(size_t)255; return p;
  };
  h16*   bufA    = (h16*)  alloc(NT*D_MODEL*2);
  h16*   bufB    = (h16*)  alloc(NT*D_MODEL*2);
  h16*   wihH    = (h16*)  alloc((size_t)2*2*192*128*2);
  h16*   whhH    = (h16*)  alloc((size_t)2*2*192*64*2);
  h16*   teoutH  = (h16*)  alloc((size_t)128*128*2);
  h16*   gatwH   = (h16*)  alloc((size_t)2*128*128*2);
  h16*   hw1H    = (h16*)  alloc((size_t)128*256*2);
  h16*   pooledH = (h16*)  alloc((size_t)N_STOCKS*128*2);
  float* tmpF    = (float*)alloc((size_t)N_STOCKS*128*4);
  float* node_emb= (float*)alloc((size_t)N_STOCKS*128*4);
  h16*   curH    = (h16*)  alloc((size_t)N_STOCKS*128*2);
  float* xl      = (float*)alloc((size_t)N_STOCKS*128*4);
  float* asrc    = (float*)alloc((size_t)N_STOCKS*4*4);
  float* adst    = (float*)alloc((size_t)N_STOCKS*4*4);
  float* ebuf    = (float*)alloc((size_t)E2*4*4);
  float* eebuf   = (float*)alloc((size_t)E2*4*4);
  unsigned* emaxu= (unsigned*)alloc((size_t)N_STOCKS*4*4);
  float* denom   = (float*)alloc((size_t)N_STOCKS*4*4);
  float* agg     = (float*)alloc((size_t)N_STOCKS*128*4);
  float* hcur    = (float*)alloc((size_t)N_STOCKS*128*4);
  float* hfin    = (float*)alloc((size_t)N_STOCKS*128*4);
  float* seq     = (float*)alloc((size_t)MEM_LEN*128*4);
  float* regret  = (float*)alloc(128*4);
  h16*   featH   = (h16*)  alloc((size_t)N_STOCKS*256*2);
  float* scores  = (float*)alloc((size_t)N_STOCKS*4);

  auto cvt = [&](const float* s, h16* d, size_t n){
    f32_to_f16_kernel<<<(unsigned)((n+255)/256), 256, 0, stream>>>(s, d, n);
  };
  cvt(te_gru_wih, wihH,  (size_t)2*2*192*128);
  cvt(te_gru_whh, whhH,  (size_t)2*2*192*64);
  cvt(te_out_w,   teoutH,(size_t)128*128);
  cvt(gat_w,      gatwH, (size_t)2*128*128);
  cvt(head_w1,    hw1H,  (size_t)128*256);

  // stage 1: input projection + LN + GELU -> bufA (f16)
  in_proj_kernel<<<(unsigned)NT, 128, 0, stream>>>(x, te_in_w, te_in_b,
                                                   te_in_ln_g, te_in_ln_b, bufA);

  // stage 2: 2-layer bidirectional GRU (in-wave WMMA scan)
  for (int l=0; l<2; ++l){
    const h16* inb = (l==0) ? bufA : bufB;
    h16* outb      = (l==0) ? bufB : bufA;
    for (int d=0; d<2; ++d){
      const size_t wo = (size_t)l*2 + d;
      gru_scan_kernel<<<(N_STOCKS/16 + 3)/4, 128, 0, stream>>>(
          inb, wihH + wo*192*128, whhH + wo*192*64,
          te_gru_bih + wo*192, te_gru_bhh + wo*192,
          outb, d, d*GRU_H);
    }
  }

  // stage 3: attention pooling + te_out projection + gelu + LN
  attnpool_kernel<<<N_STOCKS, 256, 0, stream>>>(bufA, te_attn_w, te_attn_b, pooledH);
  wmma_linear_kernel<<<dim3(N_STOCKS/16, 8), 32, 0, stream>>>(pooledH, teoutH, tmpF, 128, 128);
  gelu_ln_kernel<<<N_STOCKS, 128, 0, stream>>>(tmpF, te_out_b, te_ln_g, te_ln_b, node_emb, curH);

  // stage 4: GAT x2
  for (int l=0; l<2; ++l){
    wmma_linear_kernel<<<dim3(N_STOCKS/16, 8), 32, 0, stream>>>(
        curH, gatwH + (size_t)l*128*128, xl, 128, 128);
    gat_attn_coef_kernel<<<N_STOCKS, 128, 0, stream>>>(
        xl, gat_att_src + (size_t)l*128, gat_att_dst + (size_t)l*128, asrc, adst);
    fill_u32_kernel<<<(N_STOCKS*4+255)/256, 256, 0, stream>>>(emaxu, 0u, (size_t)N_STOCKS*4);
    fill_u32_kernel<<<(N_STOCKS*4+255)/256, 256, 0, stream>>>((unsigned*)denom, 0u, (size_t)N_STOCKS*4);
    fill_u32_kernel<<<(N_STOCKS*128+255)/256, 256, 0, stream>>>((unsigned*)agg, 0u, (size_t)N_STOCKS*128);
    gat_edge_max_kernel<<<(unsigned)(((size_t)E2*4+255)/256), 256, 0, stream>>>(
        edge_index, E, asrc, adst, ebuf, emaxu);
    gat_edge_soft_kernel<<<(unsigned)(((size_t)E2*4+255)/256), 256, 0, stream>>>(
        edge_index, E, ebuf, emaxu, eebuf, denom);
    gat_scatter_kernel<<<E2, 128, 0, stream>>>(edge_index, E, xl, eebuf, denom, agg);
    gat_finish_kernel<<<(unsigned)(((size_t)N_STOCKS*128+255)/256), 256, 0, stream>>>(
        agg, gat_bias + (size_t)l*128, hcur, curH);
  }
  add_ln_kernel<<<N_STOCKS, 128, 0, stream>>>(hcur, node_emb, gat_ln_g, gat_ln_b, hfin);

  // stage 5: portfolio memory
  mem_seq_kernel<<<MEM_LEN, 128, 0, stream>>>(history, mem_in_w, mem_in_b, seq);
  mem_gru_kernel<<<1, 128, 0, stream>>>(seq, mem_gru_wih, mem_gru_whh,
                                        mem_gru_bih, mem_gru_bhh, mem_ln_g, mem_ln_b, regret);

  // stage 6: head + softmax + projection
  feat_kernel<<<(unsigned)(((size_t)N_STOCKS*256+255)/256), 256, 0, stream>>>(hfin, regret, featH);
  wmma_linear_kernel<<<dim3(N_STOCKS/16, 8), 32, 0, stream>>>(featH, hw1H, tmpF, 128, 256);
  head_kernel<<<N_STOCKS, 128, 0, stream>>>(tmpF, head_b1, head_w2, head_b2, scores);
  softmax_project_kernel<<<1, 1024, 0, stream>>>(scores, temperature, (float*)d_out);
}
